// ResGraphConvUnpool_66743791780077
// MI455X (gfx1250) — matrix operand order repaired and checked by reading, not compiled, not run on hardware
//
#include <hip/hip_runtime.h>
#include <hip/hip_bf16.h>

#define Bsz  8
#define Npts 4096
#define DIMF 128
#define KNN  8
#define NBLK 12
#define EPSV 1e-5f

typedef __attribute__((ext_vector_type(16))) __bf16 v16bf;
typedef __attribute__((ext_vector_type(8)))  float  v8f;
typedef __attribute__((ext_vector_type(4)))  unsigned int v4u;
typedef __attribute__((ext_vector_type(8)))  int    v8i;
typedef __attribute__((ext_vector_type(4)))  int    v4i;

struct Frag32 { unsigned int u[8]; };

__device__ __forceinline__ v16bf frag_cast(Frag32 f) {
    return __builtin_bit_cast(v16bf, f);
}

// fp32 -> bf16 round-to-nearest-even, packed pair
__device__ __forceinline__ unsigned int pack_bf2(float lo, float hi) {
    unsigned int ul = __builtin_bit_cast(unsigned int, lo);
    unsigned int uh = __builtin_bit_cast(unsigned int, hi);
    unsigned int rl = (ul + 0x7FFFu + ((ul >> 16) & 1u)) >> 16;
    unsigned int rh = (uh + 0x7FFFu + ((uh >> 16) & 1u)) >> 16;
    return (rl & 0xFFFFu) | (rh << 16);
}
__device__ __forceinline__ float bf_lo(unsigned int p) { return __builtin_bit_cast(float, p << 16); }
__device__ __forceinline__ float bf_hi(unsigned int p) { return __builtin_bit_cast(float, p & 0xFFFF0000u); }

// load an 8-dword fragment from LDS with two b128 reads
__device__ __forceinline__ Frag32 ld_frag_lds(const unsigned int* p) {
    const uint4* q = (const uint4*)p;
    uint4 a = q[0], b = q[1];
    Frag32 f;
    f.u[0] = a.x; f.u[1] = a.y; f.u[2] = a.z; f.u[3] = a.w;
    f.u[4] = b.x; f.u[5] = b.y; f.u[6] = b.z; f.u[7] = b.w;
    return f;
}

// ---------------------------------------------------------------------------
// KNN: brute force, LDS-tiled candidates, per-thread register top-8
// ---------------------------------------------------------------------------
__global__ __launch_bounds__(256) void knn_kernel(const float* __restrict__ xyz,
                                                  int* __restrict__ idx) {
    __shared__ float sx[256], sy[256], sz[256];
    const int b = blockIdx.x >> 4;
    const int n = ((blockIdx.x & 15) << 8) + threadIdx.x;
    const float* xb = xyz + (size_t)b * 3 * Npts;
    const float qx = xb[n], qy = xb[Npts + n], qz = xb[2 * Npts + n];

    float bd[KNN]; int bi[KNN];
#pragma unroll
    for (int k = 0; k < KNN; ++k) { bd[k] = 3.4e38f; bi[k] = 0; }

    for (int t = 0; t < Npts; t += 256) {
        __syncthreads();
        sx[threadIdx.x] = xb[t + threadIdx.x];
        sy[threadIdx.x] = xb[Npts + t + threadIdx.x];
        sz[threadIdx.x] = xb[2 * Npts + t + threadIdx.x];
        __syncthreads();
        for (int j = 0; j < 256; ++j) {
            float dx = qx - sx[j], dy = qy - sy[j], dz = qz - sz[j];
            float d = dx * dx + dy * dy + dz * dz;
            if (d < bd[KNN - 1]) {
                bd[KNN - 1] = d; bi[KNN - 1] = t + j;
#pragma unroll
                for (int s = KNN - 1; s > 0; --s) {
                    if (bd[s] < bd[s - 1]) {
                        float td = bd[s]; bd[s] = bd[s - 1]; bd[s - 1] = td;
                        int   ti = bi[s]; bi[s] = bi[s - 1]; bi[s - 1] = ti;
                    }
                }
            }
        }
    }
    int* op = idx + ((size_t)b * Npts + n) * KNN;
#pragma unroll
    for (int k = 0; k < KNN; ++k) op[k] = bi[k];
}

// ---------------------------------------------------------------------------
// one-time fp32 -> bf16 pack of conv weights, pre-shuffled into WMMA A-frag
// order: [blk][o_tile(8)][k_tile(4)][lane(32)][8 dwords] -- each lane's
// fragment is 32 contiguous bytes, so LDS reads are 2x ds_load_b128.
// ---------------------------------------------------------------------------
__global__ __launch_bounds__(256) void wconv_kernel(const float* __restrict__ w,
                                                    unsigned int* __restrict__ w16) {
    int g = blockIdx.x * 256 + threadIdx.x;
    if (g >= NBLK * DIMF * 64) return;
    int i    = g >> 13;            // block
    int r    = g & 8191;
    int v    = r & 7;
    int lane = (r >> 3) & 31;
    int kt   = (r >> 8) & 3;
    int ot   = r >> 10;
    int m    = lane & 15, half = lane >> 4;
    int kk   = kt * 32 + half * 16 + (v << 1);
    int o    = ot * 16 + m;
    const float* src = w + (size_t)i * DIMF * DIMF + (size_t)o * DIMF + kk;
    w16[g] = pack_bf2(src[0], src[1]);
}

// ---------------------------------------------------------------------------
// BN(eval)+ReLU and transpose [B,128,N] fp32 -> hT [B,N,128] packed bf16
// ---------------------------------------------------------------------------
__global__ __launch_bounds__(256) void bn_relu_kernel(
    const float* __restrict__ pts, const float* __restrict__ gamma,
    const float* __restrict__ beta, const float* __restrict__ mean,
    const float* __restrict__ var, unsigned int* __restrict__ hT, int blk) {
    __shared__ float lds[DIMF][65];
    __shared__ float s_inv[DIMF], s_beta[DIMF], s_mean[DIMF];
    const int b  = blockIdx.x >> 6;
    const int n0 = (blockIdx.x & 63) << 6;
    const int tid = threadIdx.x;
    if (tid < DIMF) {
        float g = gamma[blk * DIMF + tid], v = var[blk * DIMF + tid];
        s_inv[tid]  = g * rsqrtf(v + EPSV);
        s_beta[tid] = beta[blk * DIMF + tid];
        s_mean[tid] = mean[blk * DIMF + tid];
    }
    __syncthreads();
#pragma unroll 4
    for (int it = 0; it < 32; ++it) {
        int l = it * 256 + tid;
        int c = l >> 6, n = l & 63;
        float p = pts[((size_t)b * DIMF + c) * Npts + n0 + n];
        float h = (p - s_mean[c]) * s_inv[c] + s_beta[c];
        lds[c][n] = h > 0.f ? h : 0.f;
    }
    __syncthreads();
#pragma unroll 4
    for (int it = 0; it < 16; ++it) {
        int l = it * 256 + tid;
        int n = l >> 6, cp = l & 63;
        hT[((size_t)b * Npts + n0 + n) * 64 + cp] = pack_bf2(lds[2 * cp][n], lds[2 * cp + 1][n]);
    }
}

// ---------------------------------------------------------------------------
// Fused: TDM-stage weights to LDS (overlapped, TENSORcnt) + 9-way neighbor
// aggregation (bf16 gather) -> LDS tile -> WMMA GEMM.
//   out = W @ (agg/9) + bias + shortcut
// One workgroup = 32 points x all 128 outputs. 8 waves, each wave does two
// 16x16 output tiles, K-loop 4 x 32 with v_wmma_f32_16x16x32_bf16.
// ---------------------------------------------------------------------------
__global__ __launch_bounds__(256) void agg_gemm_kernel(
    const unsigned int* __restrict__ hT, const unsigned int* __restrict__ w16,
    const float* __restrict__ conv_b, const int* __restrict__ idx,
    const float* __restrict__ pin, float* __restrict__ pout, int blk) {
    __shared__ unsigned int s_w[8192];        // 32KB: this block's weights, frag order
    __shared__ unsigned int s_agg[32 * 72];   // 32 points x 128 bf16 (stride 72 dwords)
    const int b   = blockIdx.x >> 7;
    const int n0  = (blockIdx.x & 127) << 5;
    const int tid = threadIdx.x;

    // ---- kick off TDM: DMA 32KB weight block -> LDS (wave 0 issues) ----
    if (tid < 32) {
        unsigned long long ga = (unsigned long long)(const void*)(w16 + (size_t)blk * 8192);
        unsigned int lds_base = (unsigned int)(size_t)&s_w[0];
        v4u g0;
        g0.x = 1u;                                            // count=1 user D#
        g0.y = lds_base;                                      // lds_addr
        g0.z = (unsigned int)(ga & 0xFFFFFFFFu);              // global_addr[31:0]
        g0.w = (unsigned int)((ga >> 32) & 0x01FFFFFFu)       // global_addr[56:32]
             | (2u << 30);                                    // type=2 (image)
        v8i g1;
        g1[0] = 0x00010000;                                   // data_size=2B
        g1[1] = (int)(16384u << 16);                          // tensor_dim0 lo16
        g1[2] = (int)(1u << 16);                              // tensor_dim1=1
        g1[3] = (int)(16384u << 16);                          // tile_dim0=16384
        g1[4] = 0;                                            // tile_dim1/2 unused
        g1[5] = 16384;                                        // tensor_dim0_stride
        g1[6] = 0;
        g1[7] = 0;
        v4i z4 = {0, 0, 0, 0};
        v8i z8 = {0, 0, 0, 0, 0, 0, 0, 0};
        __builtin_amdgcn_tensor_load_to_lds(g0, g1, z4, z4, z8, 0);
    }

    // ---- aggregation (overlaps the TDM copy): thread = (point, 16-ch seg) ----
    {
        const int n = tid >> 3, seg = tid & 7;
        const int p = n0 + n;
        const int* ip = idx + ((size_t)b * Npts + p) * KNN;
        int rows[9];
        rows[0] = p;
#pragma unroll
        for (int k = 0; k < 8; ++k) rows[1 + k] = ip[k];
        float acc[16];
#pragma unroll
        for (int k = 0; k < 16; ++k) acc[k] = 0.f;
#pragma unroll
        for (int j = 0; j < 9; ++j) {
            const uint4* hp = (const uint4*)(hT + ((size_t)(b * Npts + rows[j]) * 64) + seg * 8);
            uint4 u0 = hp[0], u1 = hp[1];
            unsigned int uu[8] = {u0.x, u0.y, u0.z, u0.w, u1.x, u1.y, u1.z, u1.w};
#pragma unroll
            for (int k = 0; k < 8; ++k) {
                acc[2 * k]     += bf_lo(uu[k]);
                acc[2 * k + 1] += bf_hi(uu[k]);
            }
        }
        const float s = 1.f / 9.f;
        unsigned int packed[8];
#pragma unroll
        for (int k = 0; k < 8; ++k) packed[k] = pack_bf2(acc[2 * k] * s, acc[2 * k + 1] * s);
        uint4* sp = (uint4*)&s_agg[n * 72 + seg * 8];
        sp[0] = make_uint4(packed[0], packed[1], packed[2], packed[3]);
        sp[1] = make_uint4(packed[4], packed[5], packed[6], packed[7]);
    }

    // ---- wait for the weight DMA, then full-workgroup barrier ----
    if (tid < 32) __builtin_amdgcn_s_wait_tensorcnt(0);
    __syncthreads();

    // ---- GEMM via WMMA; all fragments now come from LDS as 2x b128 ----
    const int lane = tid & 31, w = tid >> 5;
    const int half = lane >> 4, m = lane & 15;
    const int noff = (w & 1) * 16;        // which 16-point tile
    const int ot0  = (w >> 1) * 2;        // two 16-row output tiles per wave

    v8f c0 = {};
    v8f c1 = {};
#pragma unroll
    for (int kt = 0; kt < 4; ++kt) {
        Frag32 fa0 = ld_frag_lds(&s_w[ot0 * 1024 + kt * 256 + lane * 8]);
        Frag32 fa1 = ld_frag_lds(&s_w[(ot0 + 1) * 1024 + kt * 256 + lane * 8]);
        Frag32 fb  = ld_frag_lds(&s_agg[(noff + m) * 72 + kt * 16 + half * 8]);
        c0 = __builtin_amdgcn_wmma_f32_16x16x32_bf16(false, frag_cast(fa0), false, frag_cast(fb),
                                                     (short)0, c0, false, false);
        c1 = __builtin_amdgcn_wmma_f32_16x16x32_bf16(false, frag_cast(fa1), false, frag_cast(fb),
                                                     (short)0, c1, false, false);
    }

    // ---- fused bias + residual + store ----
    const int nout = n0 + noff + m;
#pragma unroll
    for (int r = 0; r < 8; ++r) {
        int o0 = ot0 * 16 + half * 8 + r;
        int o1 = (ot0 + 1) * 16 + half * 8 + r;
        size_t i0 = ((size_t)b * DIMF + o0) * Npts + nout;
        size_t i1 = ((size_t)b * DIMF + o1) * Npts + nout;
        pout[i0] = c0[r] + conv_b[blk * DIMF + o0] + pin[i0];
        pout[i1] = c1[r] + conv_b[blk * DIMF + o1] + pin[i1];
    }
}

// ---------------------------------------------------------------------------
// Final unpool: per point, 6 center dots (up_c_w . h) + 6 neighbor-sum dots
// new_xyz[b,d,u*N+n] = (pc+pn)/9 + xyz
// ---------------------------------------------------------------------------
__global__ __launch_bounds__(256) void unpool_kernel(
    const unsigned int* __restrict__ hT, const int* __restrict__ idx,
    const float* __restrict__ ucw, const float* __restrict__ ucb,
    const float* __restrict__ unw, const float* __restrict__ unb,
    const float* __restrict__ xyz, float* __restrict__ out) {
    const int b = blockIdx.x >> 4;
    const int n = ((blockIdx.x & 15) << 8) + threadIdx.x;
    const int* ip = idx + ((size_t)b * Npts + n) * KNN;
    float accc[6] = {0, 0, 0, 0, 0, 0}, accn[6] = {0, 0, 0, 0, 0, 0};

    {   // center row with up_c_w
        const unsigned int* hp = hT + (size_t)(b * Npts + n) * 64;
        for (int cp = 0; cp < 64; ++cp) {
            unsigned int u = hp[cp];
            float lo = bf_lo(u), hi = bf_hi(u);
#pragma unroll
            for (int o = 0; o < 6; ++o)
                accc[o] += ucw[o * DIMF + 2 * cp] * lo + ucw[o * DIMF + 2 * cp + 1] * hi;
        }
    }
#pragma unroll 1
    for (int j = 0; j < 8; ++j) {   // neighbor rows with up_n_w
        const unsigned int* hp = hT + (size_t)(b * Npts + ip[j]) * 64;
        for (int cp = 0; cp < 64; ++cp) {
            unsigned int u = hp[cp];
            float lo = bf_lo(u), hi = bf_hi(u);
#pragma unroll
            for (int o = 0; o < 6; ++o)
                accn[o] += unw[o * DIMF + 2 * cp] * lo + unw[o * DIMF + 2 * cp + 1] * hi;
        }
    }
    const float s = 1.f / 9.f;
#pragma unroll
    for (int d = 0; d < 3; ++d) {
        float base = xyz[((size_t)b * 3 + d) * Npts + n];
#pragma unroll
        for (int u = 0; u < 2; ++u) {
            int o = d * 2 + u;
            float v = (accc[o] + ucb[o] + accn[o] + 8.f * unb[o]) * s + base;
            out[(size_t)b * (3 * 2 * Npts) + (size_t)d * (2 * Npts) + (size_t)u * Npts + n] = v;
        }
    }
}

// ---------------------------------------------------------------------------
extern "C" void kernel_launch(void* const* d_in, const int* in_sizes, int n_in,
                              void* d_out, int out_size, void* d_ws, size_t ws_size,
                              hipStream_t stream) {
    const float* xyz    = (const float*)d_in[0];
    const float* points = (const float*)d_in[1];
    const float* gamma  = (const float*)d_in[2];
    const float* beta   = (const float*)d_in[3];
    const float* mean   = (const float*)d_in[4];
    const float* var    = (const float*)d_in[5];
    const float* conv_w = (const float*)d_in[6];
    const float* conv_b = (const float*)d_in[7];
    const float* up_c_w = (const float*)d_in[8];
    const float* up_c_b = (const float*)d_in[9];
    const float* up_n_w = (const float*)d_in[10];
    const float* up_n_b = (const float*)d_in[11];
    float* out = (float*)d_out;

    // workspace layout (~43.4 MB total)
    char* ws = (char*)d_ws;
    size_t off = 0;
    int* idx = (int*)(ws + off);                   off += (size_t)Bsz * Npts * KNN * 4;    // 1 MB
    unsigned int* hT  = (unsigned int*)(ws + off); off += (size_t)Bsz * Npts * DIMF * 2;   // 8 MB
    unsigned int* w16 = (unsigned int*)(ws + off); off += (size_t)NBLK * DIMF * DIMF * 2;  // 384 KB
    float* pA = (float*)(ws + off);                off += (size_t)Bsz * DIMF * Npts * 4;   // 16 MB
    float* pB = (float*)(ws + off);                off += (size_t)Bsz * DIMF * Npts * 4;   // 16 MB

    float* out_pts = out + (size_t)Bsz * 3 * 2 * Npts;   // points follow new_xyz in d_out

    knn_kernel<<<Bsz * (Npts / 256), 256, 0, stream>>>(xyz, idx);
    wconv_kernel<<<(NBLK * DIMF * 64 + 255) / 256, 256, 0, stream>>>(conv_w, w16);

    for (int i = 0; i < NBLK; ++i) {
        const float* pin  = (i == 0) ? points : ((i & 1) ? pA : pB);
        float*       pout = (i == NBLK - 1) ? out_pts : ((i & 1) ? pB : pA);
        bn_relu_kernel<<<Bsz * (Npts / 64), 256, 0, stream>>>(pin, gamma, beta, mean, var, hT, i);
        agg_gemm_kernel<<<Bsz * (Npts / 32), 256, 0, stream>>>(hT, w16, conv_b, idx, pin, pout, i);
    }
    unpool_kernel<<<Bsz * (Npts / 256), 256, 0, stream>>>(hT, idx, up_c_w, up_c_b, up_n_w, up_n_b, xyz, out);
}
